// T5Attention_71554155152048
// MI455X (gfx1250) — matrix operand order repaired and checked by reading, not compiled, not run on hardware
//
#include <hip/hip_runtime.h>

// ---------------------------------------------------------------------------
// T5 attention (bs=2, qlen=2048, d_model=512, H=8, d_kv=64) for gfx1250.
// bf16 WMMA (v_wmma_f32_16x16x32_bf16) for all GEMMs, fp32 flash softmax,
// async global->LDS staging (GLOBAL_LOAD_ASYNC_TO_LDS_B128) for K/V tiles.
// ---------------------------------------------------------------------------

typedef __attribute__((ext_vector_type(16))) __bf16 v16bf;
typedef __attribute__((ext_vector_type(8)))  float  v8f;

#define BS      2
#define QLEN    2048
#define DMODEL  512
#define NHEADS  8
#define DKV     64
#define MROWS   (BS * QLEN)   // 4096

union FragBF { v16bf v; uint4 u[2]; };

__device__ __forceinline__ unsigned short f2bf(float f) {
  unsigned int u = __float_as_uint(f);
  u += 0x7FFFu + ((u >> 16) & 1u);     // round-to-nearest-even
  return (unsigned short)(u >> 16);
}

// A-matrix fragment (16x32 bf16): 16 contiguous K elements per lane.
__device__ __forceinline__ v16bf loadA16(const unsigned short* p) {
  FragBF f;
  f.u[0] = *(const uint4*)(p);
  f.u[1] = *(const uint4*)(p + 8);
  return f.v;
}
// B-matrix fragment (32x16 bf16): 16 contiguous K elements per lane.
__device__ __forceinline__ v16bf loadB16(const unsigned short* p) {
  FragBF f;
  f.u[0] = *(const uint4*)(p);
  f.u[1] = *(const uint4*)(p + 8);
  return f.v;
}

__device__ __forceinline__ v8f wmma_bf16(v16bf a, v16bf b, v8f c) {
  return __builtin_amdgcn_wmma_f32_16x16x32_bf16(false, a, false, b,
                                                 (short)0, c, false, false);
}

// Async global->LDS 16B copy (per lane). lds_off = byte offset in workgroup LDS.
__device__ __forceinline__ void async_load_b128(unsigned lds_off,
                                                const unsigned short* gptr) {
  asm volatile("global_load_async_to_lds_b128 %0, %1, off"
               :: "v"(lds_off), "v"((unsigned long long)(size_t)gptr)
               : "memory");
}
__device__ __forceinline__ void async_wait() {
  asm volatile("s_wait_asynccnt 0x0" ::: "memory");
}

// ---------------------------------------------------------------------------
// Kernel 0a: fp32 -> bf16 elementwise convert (for X)
// ---------------------------------------------------------------------------
__global__ __launch_bounds__(256) void convert_x(const float* __restrict__ src,
                                                 unsigned short* __restrict__ dst,
                                                 int n) {
  int i = blockIdx.x * 256 + threadIdx.x;
  if (i < n) dst[i] = f2bf(src[i]);
}

// ---------------------------------------------------------------------------
// Kernel 0b: transpose + convert weights: Wt[n][k] = bf16(W[k][n])  (512x512)
// ---------------------------------------------------------------------------
__global__ __launch_bounds__(256) void transpose_w(const float* __restrict__ Wq,
                                                   const float* __restrict__ Wk,
                                                   const float* __restrict__ Wv,
                                                   const float* __restrict__ Wo,
                                                   unsigned short* __restrict__ Wqt,
                                                   unsigned short* __restrict__ Wkt,
                                                   unsigned short* __restrict__ Wvt,
                                                   unsigned short* __restrict__ Wot) {
  int which = blockIdx.z;
  const float* src = (which == 0) ? Wq : (which == 1) ? Wk : (which == 2) ? Wv : Wo;
  unsigned short* dst = (which == 0) ? Wqt : (which == 1) ? Wkt : (which == 2) ? Wvt : Wot;
  int n = blockIdx.x * 16 + (threadIdx.x & 15);
  int k = blockIdx.y * 16 + (threadIdx.x >> 4);
  dst[n * DMODEL + k] = f2bf(src[k * DMODEL + n]);
}

// ---------------------------------------------------------------------------
// Kernel 1/3: GEMM  C[M=4096][N=512] = A[M][512] * B^T   (B stored [N][K] bf16)
//   mode 0: write Q   -> Qb[(b*8+h)*2048 + q][64]        (bf16)
//   mode 1: write K   -> Kb same layout                  (bf16, key-major)
//   mode 2: write V^T -> Vt[(b*8+h)*64 + d][2048]        (bf16, dim-major)
//   mode 3: write fp32 Out[m][512]  (final output proj)
// ---------------------------------------------------------------------------
__global__ __launch_bounds__(128) void gemm_bf16(const unsigned short* __restrict__ A,
                                                 const unsigned short* __restrict__ B0,
                                                 const unsigned short* __restrict__ B1,
                                                 const unsigned short* __restrict__ B2,
                                                 unsigned short* __restrict__ Qb,
                                                 unsigned short* __restrict__ Kb,
                                                 unsigned short* __restrict__ Vt,
                                                 float* __restrict__ Out,
                                                 int modeBase) {
  const int mode = modeBase + blockIdx.z;
  const unsigned short* B = (mode == 1) ? B1 : (mode == 2) ? B2 : B0;

  const int lane = threadIdx.x & 31;
  const int wave = threadIdx.x >> 5;
  const int lm   = lane & 15;
  const int half = lane >> 4;

  const int mBase = blockIdx.x * 16;
  const int nBase = blockIdx.y * 64 + wave * 16;

  const unsigned short* aRow = A + (size_t)(mBase + lm) * DMODEL + half * 16;
  const unsigned short* bRow = B + (size_t)(nBase + lm) * DMODEL + half * 16;

  v8f acc = {0.f, 0.f, 0.f, 0.f, 0.f, 0.f, 0.f, 0.f};
#pragma unroll 4
  for (int k = 0; k < DMODEL; k += 32) {
    v16bf a = loadA16(aRow + k);
    v16bf b = loadB16(bRow + k);
    acc = wmma_bf16(a, b, acc);
  }

  const int r0 = half * 8;
  if (mode == 3) {
#pragma unroll
    for (int j = 0; j < 8; ++j) {
      int m = mBase + r0 + j;
      Out[(size_t)m * DMODEL + nBase + lm] = acc[j];
    }
  } else {
    const int n = nBase + lm;
    const int h = n >> 6, d = n & 63;
#pragma unroll
    for (int j = 0; j < 8; ++j) {
      int m  = mBase + r0 + j;
      int b_ = m >> 11;            // /2048
      int q  = m & 2047;
      unsigned short val = f2bf(acc[j]);
      size_t bh = (size_t)(b_ * NHEADS + h);
      if (mode == 0)      Qb[(bh * QLEN + q) * DKV + d] = val;
      else if (mode == 1) Kb[(bh * QLEN + q) * DKV + d] = val;
      else                Vt[(bh * DKV + d) * QLEN + q] = val;
    }
  }
}

// ---------------------------------------------------------------------------
// Kernel 2: flash attention. 4 waves/block share one (b,h); each wave owns a
// 16-row q tile. K/V tiles (32 kv) are staged once per block into double-
// buffered LDS with GLOBAL_LOAD_ASYNC_TO_LDS_B128 (overlapped with compute),
// cutting L2 re-reads 4x. Scores: 4 WMMAs; PV: 4 WMMAs; fp32 online softmax
// with analytic T5 relative-position bucket bias.
// ---------------------------------------------------------------------------
#define KSTRIDE 88   // padded bf16 elems per K row  (dword stride 44 -> no bank conflicts)
#define VSTRIDE 40   // padded bf16 elems per V row  (dword stride 20 -> no bank conflicts)

struct WaveLds {
  float S[16][33];                        // score tile, padded stride
  float fac[16];                          // per-row rescale / final 1/l
  __align__(16) unsigned short P[16][40]; // probs bf16, 80B row stride
};

struct StageLds {
  __align__(16) unsigned short K[2][32 * KSTRIDE]; // [buf][key*KSTRIDE+dim]
  __align__(16) unsigned short V[2][64 * VSTRIDE]; // [buf][dim*VSTRIDE+kv]
};

__global__ __launch_bounds__(128) void flash_attn(const unsigned short* __restrict__ Qb,
                                                  const unsigned short* __restrict__ Kb,
                                                  const unsigned short* __restrict__ Vt,
                                                  const float* __restrict__ rel_bias, // [32][8]
                                                  unsigned short* __restrict__ Ctxb)  // [4096][512]
{
  __shared__ float biasLds[32 * NHEADS];
  __shared__ WaveLds wls[4];
  __shared__ StageLds stage;

  const int tid  = threadIdx.x;
  const int lane = tid & 31;
  const int wave = tid >> 5;
  const int lm   = lane & 15;
  const int half = lane >> 4;
  const int r0   = half * 8;

  const int gid = blockIdx.x * 4 + wave;   // 0..2047
  const int bh  = gid >> 7;                // 0..15 = b*8+h (same for all 4 waves)
  const int qt  = gid & 127;
  const int h   = bh & 7;
  const int q0  = qt * 16;

  const unsigned short* kColBase = Kb + (size_t)bh * QLEN * DKV;  // + key*64
  const unsigned short* vColBase = Vt + (size_t)bh * DKV * QLEN;  // + dim*2048

  // per-thread staging chunk coordinates (2 chunks each for K and V)
  const int kr0 = tid >> 3,        ki0 = tid & 7;          // chunks tid, tid+128
  const int kr1 = (tid + 128) >> 3, ki1 = (tid + 128) & 7;
  const int vr0 = tid >> 2,        vi0 = tid & 3;
  const int vr1 = (tid + 128) >> 2, vi1 = (tid + 128) & 3;

  const unsigned kLds0 = (unsigned)(size_t)&stage.K[0][0];
  const unsigned kLds1 = (unsigned)(size_t)&stage.K[1][0];
  const unsigned vLds0 = (unsigned)(size_t)&stage.V[0][0];
  const unsigned vLds1 = (unsigned)(size_t)&stage.V[1][0];

#define STAGE_TILES(buf, kv)                                                      \
  do {                                                                            \
    unsigned kb = (buf) ? kLds1 : kLds0;                                          \
    unsigned vb = (buf) ? vLds1 : vLds0;                                          \
    async_load_b128(kb + (unsigned)(kr0 * KSTRIDE + ki0 * 8) * 2,                 \
                    kColBase + (size_t)((kv) + kr0) * DKV + ki0 * 8);             \
    async_load_b128(kb + (unsigned)(kr1 * KSTRIDE + ki1 * 8) * 2,                 \
                    kColBase + (size_t)((kv) + kr1) * DKV + ki1 * 8);             \
    async_load_b128(vb + (unsigned)(vr0 * VSTRIDE + vi0 * 8) * 2,                 \
                    vColBase + (size_t)vr0 * QLEN + (kv) + vi0 * 8);              \
    async_load_b128(vb + (unsigned)(vr1 * VSTRIDE + vi1 * 8) * 2,                 \
                    vColBase + (size_t)vr1 * QLEN + (kv) + vi1 * 8);              \
  } while (0)

  // kick off first K/V tile while we load bias + Q fragments
  STAGE_TILES(0, 0);

  for (int i = tid; i < 32 * NHEADS; i += 128) biasLds[i] = rel_bias[i];

  const unsigned short* qPtr = Qb + ((size_t)bh * QLEN + q0 + lm) * DKV + half * 16;
  v16bf qf0 = loadA16(qPtr);
  v16bf qf1 = loadA16(qPtr + 32);

  WaveLds& L = wls[wave];

  v8f ctx[4];
#pragma unroll
  for (int t = 0; t < 4; ++t) ctx[t] = (v8f){0.f,0.f,0.f,0.f,0.f,0.f,0.f,0.f};

  float rowm = -__builtin_inff();   // valid in lanes 0..15
  float rowl = 0.0f;

  async_wait();
  __syncthreads();

  int cur = 0;
  for (int kv0 = 0; kv0 < QLEN; kv0 += 32) {
    if (kv0 + 32 < QLEN) STAGE_TILES(cur ^ 1, kv0 + 32);

    const unsigned short* kTile = &stage.K[cur][0];
    const unsigned short* vTile = &stage.V[cur][0];

    // ---- scores: two 16-key tiles, K-dim 64 = 2 WMMAs each ----
#pragma unroll
    for (int t = 0; t < 2; ++t) {
      const unsigned short* kPtr = kTile + (t * 16 + lm) * KSTRIDE + half * 16;
      v8f s = {0.f, 0.f, 0.f, 0.f, 0.f, 0.f, 0.f, 0.f};
      s = wmma_bf16(qf0, loadB16(kPtr),      s);
      s = wmma_bf16(qf1, loadB16(kPtr + 32), s);
#pragma unroll
      for (int j = 0; j < 8; ++j) L.S[r0 + j][t * 16 + lm] = s[j];
    }
    __syncthreads();

    // ---- online softmax with T5 rel-pos bias (lanes 0..15, one row each) ----
    if (lane < 16) {
      const int qi = q0 + lane;
      float mblk = -__builtin_inff();
#pragma unroll 4
      for (int j = 0; j < 32; ++j) {
        int kj = kv0 + j;
        int n  = qi - kj;                 // n = -(rel_pos)
        int ret = 0;
        if (n < 0) { ret = 16; n = -n; }
        int bucket;
        if (n < 8) {
          bucket = ret + n;
        } else {
          // 8 + log(n/8)/log(16)*8 ; 8/ln(16) = 2.885390082
          int v = 8 + (int)(__logf((float)n * 0.125f) * 2.885390082f);
          bucket = ret + (v < 15 ? v : 15);
        }
        float sv = L.S[lane][j] + biasLds[bucket * NHEADS + h];
        L.S[lane][j] = sv;
        mblk = fmaxf(mblk, sv);
      }
      float newm = fmaxf(rowm, mblk);
      float f    = __expf(rowm - newm);
      float sum  = 0.f;
#pragma unroll 4
      for (int j = 0; j < 32; ++j) {
        float p = __expf(L.S[lane][j] - newm);
        sum += p;
        L.P[lane][j] = f2bf(p);
      }
      rowl = rowl * f + sum;
      rowm = newm;
      L.fac[lane] = f;
    }
    __syncthreads();

    // ---- rescale context accumulators by per-row factor ----
    float fj[8];
#pragma unroll
    for (int j = 0; j < 8; ++j) fj[j] = L.fac[r0 + j];
#pragma unroll
    for (int t = 0; t < 4; ++t)
#pragma unroll
      for (int j = 0; j < 8; ++j) ctx[t][j] *= fj[j];

    // ---- context += P(16x32) @ V(32x64): 4 dim-tiles ----
    v16bf pf = loadA16(&L.P[lm][half * 16]);
#pragma unroll
    for (int t = 0; t < 4; ++t) {
      const unsigned short* vPtr = vTile + (t * 16 + lm) * VSTRIDE + half * 16;
      ctx[t] = wmma_bf16(pf, loadB16(vPtr), ctx[t]);
    }

    // end of iteration: next tile's async copy done + all waves done reading
    async_wait();
    __syncthreads();
    cur ^= 1;
  }

  // ---- normalize by 1/l and store context bf16 [b][q][h*64+d] ----
  if (lane < 16) L.fac[lane] = (rowl > 0.f) ? (1.0f / rowl) : 0.f;
  __syncthreads();
  float inv[8];
#pragma unroll
  for (int j = 0; j < 8; ++j) inv[j] = L.fac[r0 + j];

  const int b_ = bh >> 3;
#pragma unroll
  for (int t = 0; t < 4; ++t) {
    const int dcol = h * DKV + t * 16 + lm;
#pragma unroll
    for (int j = 0; j < 8; ++j) {
      const int qrow = q0 + r0 + j;
      Ctxb[((size_t)(b_ * QLEN + qrow)) * DMODEL + dcol] = f2bf(ctx[t][j] * inv[j]);
    }
  }
#undef STAGE_TILES
}

// ---------------------------------------------------------------------------
// Host launch. Workspace layout (bf16 == unsigned short elements):
//   Xb[4096*512] | Wqt|Wkt|Wvt|Wot[512*512 each] | Qb|Kb|Vt[16*2048*64 each]
//   | Ctxb[4096*512]                              (~22 MiB total)
// ---------------------------------------------------------------------------
extern "C" void kernel_launch(void* const* d_in, const int* in_sizes, int n_in,
                              void* d_out, int out_size, void* d_ws, size_t ws_size,
                              hipStream_t stream) {
  (void)in_sizes; (void)n_in; (void)out_size; (void)ws_size;

  const float* X        = (const float*)d_in[0];
  const float* Wq       = (const float*)d_in[1];
  const float* Wk       = (const float*)d_in[2];
  const float* Wv       = (const float*)d_in[3];
  const float* Wo       = (const float*)d_in[4];
  const float* rel_bias = (const float*)d_in[5];
  float* out            = (float*)d_out;

  unsigned short* ws = (unsigned short*)d_ws;
  const size_t XB_E = (size_t)MROWS * DMODEL;          // 2,097,152
  const size_t W_E  = (size_t)DMODEL * DMODEL;         //   262,144
  const size_t HB_E = (size_t)BS * NHEADS * QLEN * DKV;// 2,097,152

  unsigned short* Xb   = ws;
  unsigned short* Wqt  = Xb  + XB_E;
  unsigned short* Wkt  = Wqt + W_E;
  unsigned short* Wvt  = Wkt + W_E;
  unsigned short* Wot  = Wvt + W_E;
  unsigned short* Qb   = Wot + W_E;
  unsigned short* Kb   = Qb  + HB_E;
  unsigned short* Vt   = Kb  + HB_E;
  unsigned short* Ctxb = Vt  + HB_E;

  // 0) fp32 -> bf16 conversions
  convert_x<<<dim3((unsigned)(XB_E / 256)), 256, 0, stream>>>(X, Xb, (int)XB_E);
  transpose_w<<<dim3(32, 32, 4), 256, 0, stream>>>(Wq, Wk, Wv, Wo,
                                                   Wqt, Wkt, Wvt, Wot);

  // 1) QKV projections (z = 0/1/2 -> Q/K/V)
  gemm_bf16<<<dim3(MROWS / 16, DMODEL / 64, 3), 128, 0, stream>>>(
      Xb, Wqt, Wkt, Wvt, Qb, Kb, Vt, out, 0);

  // 2) fused flash attention with async K/V staging + T5 rel-pos bias
  flash_attn<<<dim3((BS * NHEADS * (QLEN / 16)) / 4), 128, 0, stream>>>(
      Qb, Kb, Vt, rel_bias, Ctxb);

  // 3) output projection (fp32 store to d_out)
  gemm_bf16<<<dim3(MROWS / 16, DMODEL / 64, 1), 128, 0, stream>>>(
      Ctxb, Wot, Wot, Wot, Qb, Kb, Vt, out, 3);
}
